// LSTM_6863357739300
// MI455X (gfx1250) — compile-verified, matrix operand order checked by reading, not run on hardware
//
#include <hip/hip_runtime.h>
#include <hip/hip_bf16.h>

// ---------- types ----------
typedef __attribute__((ext_vector_type(16))) __bf16        v16bf;
typedef __attribute__((ext_vector_type(8)))  float         v8f;
typedef __attribute__((ext_vector_type(4)))  unsigned int  u32x4;
typedef __attribute__((ext_vector_type(4)))  float         f32x4;

struct P128 { u32x4 lo, hi; };   // 32 bytes -> bit_cast to v16bf

// ---------- constants ----------
#define B_TOT   2048
#define T_LEN   200
#define F_IN    128
#define H_DIM   256
#define O_DIM   6
#define K1      (F_IN + H_DIM)      // 384
#define K2      (H_DIM + H_DIM)     // 512
#define N_COLS  (4 * H_DIM)         // 1024
#define BM      32                  // batch rows per block
#define A1_STRIDE K1                // halves per row in LDS A1
#define A2_STRIDE K2                // halves per row in LDS A2
#define SMEM_U32X4 3584             // 57344 bytes: A1 (24576) + A2 (32768)

// fragments: 32x16 bf16 B-tiles, 512 ushorts each
#define W1_FRAGS ((K1/32) * (N_COLS/16))   // 768
#define W2_FRAGS ((K2/32) * (N_COLS/16))   // 1024

// ---------- helpers ----------
__device__ __forceinline__ unsigned short f2bf(float f) {
    unsigned int u = __float_as_uint(f);
    unsigned int r = u + 0x7FFFu + ((u >> 16) & 1u);   // RNE
    return (unsigned short)(r >> 16);
}
__device__ __forceinline__ unsigned int pack2(float a, float b) {
    return (unsigned int)f2bf(a) | ((unsigned int)f2bf(b) << 16);
}
__device__ __forceinline__ float sigm(float x) { return 1.0f / (1.0f + __expf(-x)); }
__device__ __forceinline__ float tanh_fast(float x) { return 2.0f / (1.0f + __expf(-2.0f * x)) - 1.0f; }

// A-fragment (16x32 bf16) from LDS, 16-bit A layout:
// lanes 0-15: rows 0-15, K 0..7 & 16..23 ; lanes 16-31: rows 0-15, K 8..15 & 24..31
__device__ __forceinline__ v16bf load_afrag(const unsigned short* A, int strideHalves,
                                            int mtile, int kt, int lane) {
    int row = mtile * 16 + (lane & 15);
    int kh  = (lane < 16) ? 0 : 8;
    const unsigned short* p = A + row * strideHalves + kt * 32 + kh;
    P128 t;
    t.lo = *(const u32x4*)(p);        // K base + 0..7
    t.hi = *(const u32x4*)(p + 16);   // K base + 16..23
    return __builtin_bit_cast(v16bf, t);
}

// B-fragment (32x16 bf16) pre-packed so each lane reads 32 contiguous bytes
__device__ __forceinline__ v16bf load_bfrag(const unsigned short* Wf, int fragIdx, int lane) {
    const u32x4* p = (const u32x4*)(Wf + (size_t)fragIdx * 512) + lane * 2;
    P128 t;
    t.lo = p[0];
    t.hi = p[1];
    return __builtin_bit_cast(v16bf, t);
}

__device__ __forceinline__ v8f wmma_bf16(v16bf a, v16bf b, v8f c) {
    return __builtin_amdgcn_wmma_f32_16x16x32_bf16(false, a, false, b, (short)0, c, false, false);
}

// ---------- weight repack: fp32 row-major [K][1024] -> bf16 fragment layout ----------
__global__ void pack_weights(const float* __restrict__ W, unsigned short* __restrict__ Wf,
                             int K) {
    const int N = N_COLS;
    int tid = blockIdx.x * blockDim.x + threadIdx.x;
    int nFrags = (K / 32) * (N / 16);
    if (tid >= nFrags * 32) return;
    int frag = tid >> 5;
    int lane = tid & 31;
    int kt = frag / (N / 16);
    int nt = frag % (N / 16);
    int n  = nt * 16 + (lane & 15);
    int kbase = kt * 32 + ((lane < 16) ? 0 : 8);
    unsigned short* outp = Wf + (size_t)frag * 512 + lane * 16;
#pragma unroll
    for (int h = 0; h < 16; ++h) {
        int v = h >> 1, pb = h & 1;
        int kk = (v < 4 ? 2 * v : 16 + 2 * (v - 4)) + pb;
        int k = kbase + kk;
        outp[h] = f2bf(W[(size_t)k * N + n]);
    }
}

// ---------- main persistent-time-loop LSTM kernel ----------
__global__ __launch_bounds__(512, 1)
void lstm_kernel(const float* __restrict__ x, const int* __restrict__ lengths,
                 const unsigned short* __restrict__ W1f, const unsigned short* __restrict__ W2f,
                 const float* __restrict__ b1, const float* __restrict__ b2,
                 const float* __restrict__ Wout, const float* __restrict__ bout,
                 float* __restrict__ out) {
    __shared__ u32x4 smem4[SMEM_U32X4];
    unsigned short* A1 = (unsigned short*)smem4;                 // [32][384] bf16: [x | h1]
    unsigned short* A2 = (unsigned short*)smem4 + BM * A1_STRIDE;// [32][512] bf16: [nh1 | h2]
    float* AVG = (float*)smem4;                                  // [32][256] f32 (reused)

    const int tid  = threadIdx.x;
    const int lane = tid & 31;
    const int wave = tid >> 5;              // 0..15, owns hidden cols [16w,16w+16)
    const int baseRow = blockIdx.x * BM;

    // zero LDS (h regions must start at 0)
    for (int i = tid; i < SMEM_U32X4; i += 512) smem4[i] = (u32x4)0u;

    // biases for this wave's columns (per gate), splat-ready per lane
    float bias1[4], bias2[4];
#pragma unroll
    for (int g = 0; g < 4; ++g) {
        int col = g * H_DIM + wave * 16 + (lane & 15);
        bias1[g] = b1[col];
        bias2[g] = b2[col];
    }
    // sequence lengths for the rows this lane's C-tile components map to
    int lenr[2][8];
#pragma unroll
    for (int m = 0; m < 2; ++m)
#pragma unroll
        for (int r = 0; r < 8; ++r)
            lenr[m][r] = lengths[baseRow + m * 16 + r + ((lane >= 16) ? 8 : 0)];

    v8f c1[2] = {}, c2[2] = {}, sum2[2] = {};

    // per-thread source pointer for the x staging (advances by F_IN each step)
    const int xr  = tid >> 4;            // 0..31
    const int xc0 = (tid & 15) * 8;      // 0..120
    const float* xp = x + (size_t)(baseRow + xr) * T_LEN * F_IN + xc0;

    __syncthreads();

#pragma clang loop unroll(disable)
    for (int t = 0; t < T_LEN; ++t) {
        // ---- stage 1: x_t -> bf16 into A1[:, 0:128] ----
        {
            f32x4 f0 = *(const f32x4*)(xp);
            f32x4 f1 = *(const f32x4*)(xp + 4);
            xp += F_IN;
            u32x4 q;
            q.x = pack2(f0.x, f0.y); q.y = pack2(f0.z, f0.w);
            q.z = pack2(f1.x, f1.y); q.w = pack2(f1.z, f1.w);
            *(u32x4*)(A1 + xr * A1_STRIDE + xc0) = q;
        }
        __syncthreads();

        // ---- stage 2: layer-1 GEMM  z1 = [x_t|h1] @ W1 + b1 ----
        {
            v8f acc[4][2];
#pragma unroll
            for (int g = 0; g < 4; ++g)
#pragma unroll
                for (int m = 0; m < 2; ++m) acc[g][m] = (v8f)bias1[g];

            // keep rolled: addresses stay kt-variant so weight loads cannot be
            // hoisted out of the t-loop (which previously caused scratch spills)
#pragma unroll 2
            for (int kt = 0; kt < K1 / 32; ++kt) {
                v16bf a0 = load_afrag(A1, A1_STRIDE, 0, kt, lane);
                v16bf a1 = load_afrag(A1, A1_STRIDE, 1, kt, lane);
#pragma unroll
                for (int g = 0; g < 4; ++g) {
                    int ntile = g * 16 + wave;
                    v16bf bfr = load_bfrag(W1f, kt * (N_COLS / 16) + ntile, lane);
                    acc[g][0] = wmma_bf16(a0, bfr, acc[g][0]);
                    acc[g][1] = wmma_bf16(a1, bfr, acc[g][1]);
                }
            }
            __syncthreads();   // all LDS A1 reads done before h1 rewrite

            // ---- stage 3: layer-1 activations; write raw nh1 to A2, masked h1 to A1 ----
#pragma unroll
            for (int m = 0; m < 2; ++m)
#pragma unroll
                for (int r = 0; r < 8; ++r) {
                    float iv = acc[0][m][r], jv = acc[1][m][r];
                    float fv = acc[2][m][r], ov = acc[3][m][r];
                    float cold = c1[m][r];
                    float ncv = cold * sigm(fv + 1.0f) + sigm(iv) * tanh_fast(jv);
                    float nhv = tanh_fast(ncv) * sigm(ov);
                    bool valid = t < lenr[m][r];
                    c1[m][r] = valid ? ncv : cold;
                    int row = m * 16 + r + ((lane >= 16) ? 8 : 0);
                    int col = wave * 16 + (lane & 15);
                    unsigned short nhb = f2bf(nhv);
                    A2[row * A2_STRIDE + col] = nhb;                       // raw nh1 (layer-2 input)
                    unsigned short oldb = A1[row * A1_STRIDE + F_IN + col];
                    A1[row * A1_STRIDE + F_IN + col] = valid ? nhb : oldb; // masked h1 state
                }
        }
        __syncthreads();

        // ---- stage 4: layer-2 GEMM  z2 = [nh1|h2] @ W2 + b2 ----
        {
            v8f acc[4][2];
#pragma unroll
            for (int g = 0; g < 4; ++g)
#pragma unroll
                for (int m = 0; m < 2; ++m) acc[g][m] = (v8f)bias2[g];

#pragma unroll 2
            for (int kt = 0; kt < K2 / 32; ++kt) {
                v16bf a0 = load_afrag(A2, A2_STRIDE, 0, kt, lane);
                v16bf a1 = load_afrag(A2, A2_STRIDE, 1, kt, lane);
#pragma unroll
                for (int g = 0; g < 4; ++g) {
                    int ntile = g * 16 + wave;
                    v16bf bfr = load_bfrag(W2f, kt * (N_COLS / 16) + ntile, lane);
                    acc[g][0] = wmma_bf16(a0, bfr, acc[g][0]);
                    acc[g][1] = wmma_bf16(a1, bfr, acc[g][1]);
                }
            }
            __syncthreads();   // all LDS A2 reads done before h2 rewrite

            // ---- stage 5: layer-2 activations; masked h2 to A2; accumulate pooled output ----
#pragma unroll
            for (int m = 0; m < 2; ++m)
#pragma unroll
                for (int r = 0; r < 8; ++r) {
                    float iv = acc[0][m][r], jv = acc[1][m][r];
                    float fv = acc[2][m][r], ov = acc[3][m][r];
                    float cold = c2[m][r];
                    float ncv = cold * sigm(fv + 1.0f) + sigm(iv) * tanh_fast(jv);
                    float nhv = tanh_fast(ncv) * sigm(ov);
                    bool valid = t < lenr[m][r];
                    c2[m][r] = valid ? ncv : cold;
                    int row = m * 16 + r + ((lane >= 16) ? 8 : 0);
                    int col = wave * 16 + (lane & 15);
                    unsigned short nhb = f2bf(nhv);
                    unsigned short oldb = A2[row * A2_STRIDE + H_DIM + col];
                    A2[row * A2_STRIDE + H_DIM + col] = valid ? nhb : oldb;
                    sum2[m][r] += valid ? nhv : 0.0f;
                }
        }
        // next-iteration stage-1 barrier protects the h2 writes
    }

    // ---- mean-pool + projection ----
    __syncthreads();  // done with A1/A2; reuse LDS as fp32 AVG
#pragma unroll
    for (int m = 0; m < 2; ++m)
#pragma unroll
        for (int r = 0; r < 8; ++r) {
            int row = m * 16 + r + ((lane >= 16) ? 8 : 0);
            int col = wave * 16 + (lane & 15);
            AVG[row * H_DIM + col] = sum2[m][r] / (float)lenr[m][r];
        }
    __syncthreads();

    for (int idx = tid; idx < BM * O_DIM; idx += 512) {
        int r = idx / O_DIM;
        int o = idx - r * O_DIM;
        float a = bout[o];
        for (int k = 0; k < H_DIM; ++k) a += AVG[r * H_DIM + k] * Wout[k * O_DIM + o];
        out[(size_t)(baseRow + r) * O_DIM + o] = a;
    }
}

// ---------- launch ----------
extern "C" void kernel_launch(void* const* d_in, const int* in_sizes, int n_in,
                              void* d_out, int out_size, void* d_ws, size_t ws_size,
                              hipStream_t stream) {
    (void)in_sizes; (void)n_in; (void)out_size; (void)ws_size;
    const float* x       = (const float*)d_in[0];
    const int*   lengths = (const int*)  d_in[1];
    const float* W1      = (const float*)d_in[2];
    const float* b1      = (const float*)d_in[3];
    const float* W2      = (const float*)d_in[4];
    const float* b2      = (const float*)d_in[5];
    const float* Wout    = (const float*)d_in[6];
    const float* bout    = (const float*)d_in[7];

    unsigned short* W1f = (unsigned short*)d_ws;
    unsigned short* W2f = W1f + (size_t)W1_FRAGS * 512;

    pack_weights<<<(W1_FRAGS * 32 + 255) / 256, 256, 0, stream>>>(W1, W1f, K1);
    pack_weights<<<(W2_FRAGS * 32 + 255) / 256, 256, 0, stream>>>(W2, W2f, K2);

    lstm_kernel<<<B_TOT / BM, 512, 0, stream>>>(x, lengths, W1f, W2f,
                                                b1, b2, Wout, bout, (float*)d_out);
}
